// EdgeTypeEPSSClassifier_72834055406014
// MI455X (gfx1250) — matrix operand order repaired and checked by reading, not compiled, not run on hardware
//
#include <hip/hip_runtime.h>
#include <hip/hip_bf16.h>

// ----------------------------------------------------------------------------
// EdgeTypeEPSSClassifier for MI455X (gfx1250, wave32, WMMA)
//
// All big GEMMs run as bf16 v_wmma_f32_16x16x32_bf16 with fp32 accumulation.
// Weights are pre-transposed to [N][K] bf16 so each B fragment is two aligned
// 16B loads per lane. The edge message MLP is fully fused per 16-edge tile:
// GEMM1 -> GELU -> LDS (wave-private transpose) -> GEMM2 -> f32 atomic
// scatter-add into aggr. Inner K-loops are rolled, double-buffered, and pinned
// with sched_group_barrier to [loads x10][wmma x4][loads x10][wmma x4] so the
// waitcnt pass emits partial s_wait_loadcnt and WMMA overlaps in-flight loads.
// ----------------------------------------------------------------------------

typedef __attribute__((ext_vector_type(16))) __bf16 v16bf;
typedef __attribute__((ext_vector_type(8)))  __bf16 v8bf;
typedef __attribute__((ext_vector_type(8)))  float  v8f;

// Scheduler group pinning (no-op if builtin unavailable).
// Masks: 0x008 = MFMA/WMMA, 0x020 = VMEM read, 0x100 = DS read.
#if defined(__has_builtin)
#if __has_builtin(__builtin_amdgcn_sched_group_barrier)
#define SCHED_GB(mask, size, id) __builtin_amdgcn_sched_group_barrier(mask, size, id)
#endif
#endif
#ifndef SCHED_GB
#define SCHED_GB(mask, size, id)
#endif

__device__ __forceinline__ v8f v8f_zero() {
    v8f z = {0.f,0.f,0.f,0.f,0.f,0.f,0.f,0.f};
    return z;
}

__device__ __forceinline__ v16bf cat8(v8bf lo, v8bf hi) {
    return __builtin_shufflevector(lo, hi, 0,1,2,3,4,5,6,7,8,9,10,11,12,13,14,15);
}

// A-fragment loader: per ISA 16-bit A layout, lane holds row (lane&15) with
// K chunks {base..base+7, base+16..base+23}, base = 8*(lane>=16).
// p must already point at (row base + kchunk base); 16B aligned.
__device__ __forceinline__ v16bf load_a16(const __bf16* p) {
    v8bf lo = *(const v8bf*)(p);
    v8bf hi = *(const v8bf*)(p + 16);
    return cat8(lo, hi);
}

__device__ __forceinline__ v16bf load_b32e(const __bf16* p) {
    return cat8(*(const v8bf*)(p), *(const v8bf*)(p + 16));   // 16B aligned, K chunks {base..base+7, base+16..base+23}
}

__device__ __forceinline__ v8f wmma_bf16(v16bf a, v16bf b, v8f c) {
    return __builtin_amdgcn_wmma_f32_16x16x32_bf16(false, a, false, b, (short)0, c,
                                                   false, false);
}

__device__ __forceinline__ float gelu_exact(float x) {
    return 0.5f * x * (1.0f + erff(x * 0.70710678118654752440f));
}

// --------------------------------------------------------------- GEMM panel
// Rolled, double-buffered K-segment:
//   acc[0..NT-1] += A[16, nk*32] @ B[nk*32, NT*16]
//   a0   : lane's A row pointer at segment k=0, already + (hf<<3)
//   b0   : Wt + (col0)*bstr + kglob + (hf<<3)   (col0 = n0 + r)
//   bstr : Wt row stride in elements (= K of the full GEMM)
//   nk   : number of 32-wide k-steps in this segment (must be even)
// Each iteration covers two k-steps (P then Q); prefetch is unconditional
// (final over-read <= 128B lands in adjacent workspace / OOB-LDS-returns-0 and
// is never accumulated), keeping the body branch-free so the sched groups pin
// an exact [loads][wmma][loads][wmma] software pipeline.
template <int NT, bool LDSA>
__device__ __forceinline__ void gemm_seg(v8f (&acc)[NT],
                                         const __bf16* a0,
                                         const __bf16* b0,
                                         size_t bstr, int nk) {
    v16bf aP = load_a16(a0);
    v16bf bP[NT];
#pragma unroll
    for (int t = 0; t < NT; ++t) bP[t] = load_b32e(b0 + (size_t)t * 16 * bstr);

    const int npair = nk >> 1;
#pragma clang loop unroll(disable)
    for (int i = 0; i < npair; ++i) {
        const int k0 = i * 64;
        // prefetch Q (k0+32)
        v16bf aQ = load_a16(a0 + k0 + 32);
        v16bf bQ[NT];
#pragma unroll
        for (int t = 0; t < NT; ++t)
            bQ[t] = load_b32e(b0 + (size_t)t * 16 * bstr + k0 + 32);
        // compute P
#pragma unroll
        for (int t = 0; t < NT; ++t) acc[t] = wmma_bf16(aP, bP[t], acc[t]);
        // prefetch next P (k0+64); unconditional (see note above)
        aP = load_a16(a0 + k0 + 64);
#pragma unroll
        for (int t = 0; t < NT; ++t)
            bP[t] = load_b32e(b0 + (size_t)t * 16 * bstr + k0 + 64);
        // compute Q
#pragma unroll
        for (int t = 0; t < NT; ++t) acc[t] = wmma_bf16(aQ, bQ[t], acc[t]);

        // Pin schedule: loads above the WMMAs that do not consume them.
        if (LDSA) {
            SCHED_GB(0x100, 2, 0);            // aQ (ds reads)
            SCHED_GB(0x020, 2 * NT, 0);       // bQ
            SCHED_GB(0x008, NT, 0);           // wmma P
            SCHED_GB(0x100, 2, 0);            // aP'
            SCHED_GB(0x020, 2 * NT, 0);       // bP'
            SCHED_GB(0x008, NT, 0);           // wmma Q
        } else {
            SCHED_GB(0x020, 2 + 2 * NT, 0);   // aQ + bQ
            SCHED_GB(0x008, NT, 0);           // wmma P
            SCHED_GB(0x020, 2 + 2 * NT, 0);   // aP' + bP'
            SCHED_GB(0x008, NT, 0);           // wmma Q
        }
    }
}

// ---------------------------------------------------------------- elementwise
__global__ void cvt_f32_bf16_kernel(const float* __restrict__ src,
                                    __bf16* __restrict__ dst, int n) {
    int i = blockIdx.x * blockDim.x + threadIdx.x;
    if (i < n) dst[i] = (__bf16)src[i];
}

__global__ void zero_f32_kernel(float* __restrict__ p, int n) {
    int i = blockIdx.x * blockDim.x + threadIdx.x;
    if (i < n) p[i] = 0.f;
}

// W[K][N] fp32 row-major  ->  Wt[N][K] bf16 (B-fragment friendly)
__global__ void wt_kernel(const float* __restrict__ W, __bf16* __restrict__ Wt,
                          int K, int N) {
    int idx = blockIdx.x * blockDim.x + threadIdx.x;
    if (idx >= K * N) return;
    int k = idx / N, n = idx % N;
    Wt[(size_t)n * K + k] = (__bf16)W[(size_t)k * N + n];
}

// ----------------------------------------------------------------- node GEMM
// out[M,256] = A[M,K] @ W[K,256] + bias   (A bf16, W pre-transposed bf16)
// CAT=1: A = concat(A0[M,256], A1[M,256]) along K (K must be 512); the K-loop
// is split per segment so the inner loop has a fixed base (no cndmask chains).
// Block: 256 threads = 8 waves = 2 M-tiles x 4 N-quarters -> 32x256 per block.
template <int CAT>
__global__ void gemm_node_kernel(const __bf16* __restrict__ A0,
                                 const __bf16* __restrict__ A1,
                                 const __bf16* __restrict__ Wt,   // [256][K]
                                 const float*  __restrict__ bias,
                                 float* __restrict__ out, int M, int K) {
    const int w    = threadIdx.x >> 5;
    const int lane = threadIdx.x & 31;
    const int r    = lane & 15;
    const int hf   = lane >> 4;            // half-wave select
    const int mbase = blockIdx.x * 32 + (w >> 2) * 16;
    const int n0    = (w & 3) * 64;

    int row = mbase + r;
    if (row >= M) row = M - 1;             // clamp loads; stores are guarded

    v8f acc[4];
#pragma unroll
    for (int t = 0; t < 4; ++t) acc[t] = v8f_zero();

    const __bf16* bcol = Wt + (size_t)(n0 + r) * K + (hf << 3);
    if (CAT) {
        gemm_seg<4,false>(acc, A0 + (size_t)row * 256 + (hf << 3), bcol,        K, 8);
        gemm_seg<4,false>(acc, A1 + (size_t)row * 256 + (hf << 3), bcol + 256,  K, 8);
    } else {
        gemm_seg<4,false>(acc, A0 + (size_t)row * K + (hf << 3),   bcol,        K, K >> 5);
    }

#pragma unroll
    for (int t = 0; t < 4; ++t) {
        int col  = n0 + t * 16 + r;
        float bv = bias[col];
#pragma unroll
        for (int rr = 0; rr < 8; ++rr) {
            int m = mbase + rr + (hf << 3);
            if (m < M) out[(size_t)m * 256 + col] = acc[t][rr] + bv;
        }
    }
}

// ------------------------------------------------------------- edge messages
// Per 16-edge tile (one wave): A = cat(h[dst],h[src],ee[etype]) [16,768] bf16
//   m1 = gelu(A @ W1 + b1)  -> LDS (bf16, wave-private transpose C->A layout)
//   m2 = m1 @ W2 + b2       -> atomicAdd into aggr[dst]
// Block = 128 threads = 4 tiles. N processed in four 64-col quarters to cap
// live accumulators at 4 x v8f (no spills).
__global__ void edge_msg_kernel(const __bf16* __restrict__ hbf,     // [N][256]
                                const __bf16* __restrict__ eebf,    // [T][256]
                                const __bf16* __restrict__ w1t,     // [256][768]
                                const float*  __restrict__ b1,
                                const __bf16* __restrict__ w2t,     // [256][256]
                                const float*  __restrict__ b2,
                                const int*    __restrict__ ei,      // [2E]
                                const int*    __restrict__ etype,   // [E]
                                float* __restrict__ aggr,           // [N][256]
                                int E) {
    __shared__ alignas(32) __bf16 lds[4][16][256];

    const int w    = threadIdx.x >> 5;
    const int lane = threadIdx.x & 31;
    const int tile = blockIdx.x * 4 + w;
    const int ntiles = E >> 4;             // E assumed multiple of 16 (200000 is)
    if (tile >= ntiles) return;

    const int e0 = tile * 16;
    const int r  = lane & 15;
    const int hf = lane >> 4;
    const int er = e0 + r;

    const int srcI = ei[er];               // edge_index[0] = src (x_j)
    const int dstI = ei[E + er];           // edge_index[1] = dst (x_i)
    const int et   = etype[er];

    // concat order: [h[dst], h[src], ee[et]]
    const __bf16* seg0 = hbf + (size_t)dstI * 256 + (hf << 3);
    const __bf16* seg1 = hbf + (size_t)srcI * 256 + (hf << 3);
    const __bf16* seg2 = eebf + (size_t)et * 256 + (hf << 3);

    // dst index of each output row this lane touches in the scatter epilogue
    int dstm[8];
#pragma unroll
    for (int rr = 0; rr < 8; ++rr) dstm[rr] = ei[E + e0 + rr + (hf << 3)];

    // ---- GEMM1: [16,768] x [768,256] -> gelu -> LDS bf16 (4 col-quarters)
    for (int nh = 0; nh < 4; ++nh) {
        v8f acc[4];
#pragma unroll
        for (int t = 0; t < 4; ++t) acc[t] = v8f_zero();

        const __bf16* bcol = w1t + (size_t)(nh * 64 + r) * 768 + (hf << 3);
        gemm_seg<4,false>(acc, seg0, bcol,       768, 8);
        gemm_seg<4,false>(acc, seg1, bcol + 256, 768, 8);
        gemm_seg<4,false>(acc, seg2, bcol + 512, 768, 8);

#pragma unroll
        for (int t = 0; t < 4; ++t) {
            int col  = nh * 64 + t * 16 + r;
            float bv = b1[col];
#pragma unroll
            for (int rr = 0; rr < 8; ++rr) {
                int m = rr + (hf << 3);
                lds[w][m][col] = (__bf16)gelu_exact(acc[t][rr] + bv);
            }
        }
    }
    // wave-private LDS region: in-wave DScnt ordering suffices, no barrier.

    // ---- GEMM2: [16,256] x [256,256] -> + b2 -> atomic scatter to aggr[dst]
    for (int nh = 0; nh < 4; ++nh) {
        v8f acc[4];
#pragma unroll
        for (int t = 0; t < 4; ++t) acc[t] = v8f_zero();

        gemm_seg<4,true>(acc, &lds[w][r][hf << 3],
                         w2t + (size_t)(nh * 64 + r) * 256 + (hf << 3), 256, 8);

#pragma unroll
        for (int t = 0; t < 4; ++t) {
            int col  = nh * 64 + t * 16 + r;
            float bv = b2[col];
#pragma unroll
            for (int rr = 0; rr < 8; ++rr) {
                atomicAdd(aggr + (size_t)dstm[rr] * 256 + col, acc[t][rr] + bv);
            }
        }
    }
}

// ------------------------------------------------------------------ layernorm
// mode 0 (input):  y = gelu(ln(tmp))
// mode 1 (layer):  y = ln(h + gelu(tmp))
// One wave per row (H=256 -> 8 channels/lane), writes h fp32 + hbf bf16.
__global__ void ln_kernel(const float* __restrict__ tmp,
                          const float* __restrict__ gamma,
                          const float* __restrict__ beta,
                          float* __restrict__ h, __bf16* __restrict__ hbf,
                          int M, int mode) {
    const int lane = threadIdx.x & 31;
    const int row  = blockIdx.x * 8 + (threadIdx.x >> 5);
    if (row >= M) return;

    float v[8];
    float s = 0.f, sq = 0.f;
#pragma unroll
    for (int j = 0; j < 8; ++j) {
        int c = lane + 32 * j;
        float x = tmp[(size_t)row * 256 + c];
        if (mode) x = h[(size_t)row * 256 + c] + gelu_exact(x);
        v[j] = x;
        s += x;
        sq += x * x;
    }
#pragma unroll
    for (int m = 16; m >= 1; m >>= 1) {
        s  += __shfl_xor(s,  m, 32);
        sq += __shfl_xor(sq, m, 32);
    }
    float mean = s * (1.f / 256.f);
    float var  = sq * (1.f / 256.f) - mean * mean;
    float rstd = rsqrtf(var + 1e-5f);
#pragma unroll
    for (int j = 0; j < 8; ++j) {
        int c = lane + 32 * j;
        float y = (v[j] - mean) * rstd * gamma[c] + beta[c];
        if (!mode) y = gelu_exact(y);
        h[(size_t)row * 256 + c]   = y;
        hbf[(size_t)row * 256 + c] = (__bf16)y;
    }
}

// ---------------------------------------------------------------- graph pool
// batch is sorted; block g binary-searches its [start,end) row range.
__global__ void pool_kernel(const float* __restrict__ h,
                            const int* __restrict__ batch,
                            float* __restrict__ g, int Nn) {
    const int gr = blockIdx.x;
    const int c  = threadIdx.x;           // 256 channels

    int lo = 0, hi = Nn;
    while (lo < hi) { int mid = (lo + hi) >> 1; if (batch[mid] < gr) lo = mid + 1; else hi = mid; }
    const int start = lo;
    hi = Nn;
    while (lo < hi) { int mid = (lo + hi) >> 1; if (batch[mid] < gr + 1) lo = mid + 1; else hi = mid; }
    const int end = lo;

    float s = 0.f, mx = -3.402823466e38f;
    for (int n = start; n < end; ++n) {
        float x = h[(size_t)n * 256 + c];
        s += x;
        mx = fmaxf(mx, x);
    }
    float cnt = (float)(end - start);
    g[(size_t)gr * 512 + c]       = s / fmaxf(cnt, 1.f);
    g[(size_t)gr * 512 + 256 + c] = mx;
}

// -------------------------------------------------------------- tiny FC head
__global__ void fc_kernel(const float* __restrict__ in, const float* __restrict__ W,
                          const float* __restrict__ b, float* __restrict__ out,
                          int M, int K, int Nout, int relu) {
    int idx = blockIdx.x * blockDim.x + threadIdx.x;
    if (idx >= M * Nout) return;
    int m = idx / Nout, n = idx % Nout;
    float s = b[n];
    for (int k = 0; k < K; ++k) s += in[(size_t)m * K + k] * W[(size_t)k * Nout + n];
    if (relu) s = fmaxf(s, 0.f);
    out[idx] = s;
}

// ============================================================================
extern "C" void kernel_launch(void* const* d_in, const int* in_sizes, int n_in,
                              void* d_out, int out_size, void* d_ws, size_t ws_size,
                              hipStream_t stream) {
    (void)in_sizes; (void)n_in; (void)out_size; (void)ws_size;

    const int Nn = 50000, Ee = 200000, INF = 512, Hh = 256, Gg = 64, Tt = 13;

    // ---- inputs: setup_inputs() dict order, params flattened depth-first ----
    const float* x     = (const float*)d_in[0];
    const int*   ei    = (const int*)d_in[1];   // [2][E]
    const int*   etyp  = (const int*)d_in[2];   // [E]
    const int*   batch = (const int*)d_in[3];   // [N], sorted
    int p = 4;
    const float* in_w    = (const float*)d_in[p++];
    const float* in_b    = (const float*)d_in[p++];
    const float* in_ln_s = (const float*)d_in[p++];
    const float* in_ln_b = (const float*)d_in[p++];
    const float *ee[3], *w1[3], *b1[3], *w2[3], *b2[3], *uw[3], *ub[3], *ls[3], *lb[3];
    for (int l = 0; l < 3; ++l) {
        ee[l] = (const float*)d_in[p++];
        w1[l] = (const float*)d_in[p++];
        b1[l] = (const float*)d_in[p++];
        w2[l] = (const float*)d_in[p++];
        b2[l] = (const float*)d_in[p++];
        uw[l] = (const float*)d_in[p++];
        ub[l] = (const float*)d_in[p++];
        ls[l] = (const float*)d_in[p++];
        lb[l] = (const float*)d_in[p++];
    }
    const float* cw1 = (const float*)d_in[p++];
    const float* cb1 = (const float*)d_in[p++];
    const float* cw2 = (const float*)d_in[p++];
    const float* cb2 = (const float*)d_in[p++];
    const float* cw3 = (const float*)d_in[p++];
    const float* cb3 = (const float*)d_in[p++];

    // ---- workspace carve (deterministic, ~205MB) ----
    char* ws = (char*)d_ws;
    size_t off = 0;
    auto take = [&](size_t bytes) -> char* {
        char* ptr = ws + off;
        off = (off + bytes + 255) & ~(size_t)255;
        return ptr;
    };
    __bf16* xbf    = (__bf16*)take((size_t)Nn * INF * 2);
    __bf16* hbf    = (__bf16*)take((size_t)Nn * Hh * 2);
    float*  h      = (float*) take((size_t)Nn * Hh * 4);
    float*  aggr   = (float*) take((size_t)Nn * Hh * 4);  // also GEMM tmp (aliased:
    float*  tmp    = aggr;                                 // aggr consumed into aggrbf first)
    __bf16* aggrbf = (__bf16*)take((size_t)Nn * Hh * 2);
    __bf16* inwt   = (__bf16*)take((size_t)INF * Hh * 2);
    __bf16 *eebf[3], *w1t[3], *w2t[3], *uwt[3];
    for (int l = 0; l < 3; ++l) {
        eebf[l] = (__bf16*)take((size_t)Tt * Hh * 2);
        w1t[l]  = (__bf16*)take((size_t)3 * Hh * Hh * 2);   // [256][768]
        w2t[l]  = (__bf16*)take((size_t)Hh * Hh * 2);       // [256][256]
        uwt[l]  = (__bf16*)take((size_t)2 * Hh * Hh * 2);   // [256][512]
    }
    float* gpool = (float*)take((size_t)Gg * 512 * 4);
    float* g1    = (float*)take((size_t)Gg * 256 * 4);
    float* g2    = (float*)take((size_t)Gg * 128 * 4);

    dim3 b256(256);
    auto gsz = [](long n) { return dim3((unsigned)((n + 255) / 256)); };

    // ---- precision conversion / weight transposes ----
    cvt_f32_bf16_kernel<<<gsz((long)Nn * INF), b256, 0, stream>>>(x, xbf, Nn * INF);
    wt_kernel<<<gsz((long)INF * Hh), b256, 0, stream>>>(in_w, inwt, INF, Hh);
    for (int l = 0; l < 3; ++l) {
        cvt_f32_bf16_kernel<<<gsz((long)Tt * Hh), b256, 0, stream>>>(ee[l], eebf[l], Tt * Hh);
        wt_kernel<<<gsz((long)3 * Hh * Hh), b256, 0, stream>>>(w1[l], w1t[l], 3 * Hh, Hh);
        wt_kernel<<<gsz((long)Hh * Hh),     b256, 0, stream>>>(w2[l], w2t[l], Hh, Hh);
        wt_kernel<<<gsz((long)2 * Hh * Hh), b256, 0, stream>>>(uw[l], uwt[l], 2 * Hh, Hh);
    }

    // ---- input projection + LN + GELU ----
    gemm_node_kernel<0><<<dim3((Nn + 31) / 32), b256, 0, stream>>>(
        xbf, (const __bf16*)nullptr, inwt, in_b, tmp, Nn, INF);
    ln_kernel<<<dim3((Nn + 7) / 8), b256, 0, stream>>>(
        tmp, in_ln_s, in_ln_b, h, hbf, Nn, 0);

    // ---- GNN layers ----
    for (int l = 0; l < 3; ++l) {
        zero_f32_kernel<<<gsz((long)Nn * Hh), b256, 0, stream>>>(aggr, Nn * Hh);
        edge_msg_kernel<<<dim3((Ee / 16 + 3) / 4), dim3(128), 0, stream>>>(
            hbf, eebf[l], w1t[l], b1[l], w2t[l], b2[l], ei, etyp, aggr, Ee);
        cvt_f32_bf16_kernel<<<gsz((long)Nn * Hh), b256, 0, stream>>>(aggr, aggrbf, Nn * Hh);
        gemm_node_kernel<1><<<dim3((Nn + 31) / 32), b256, 0, stream>>>(
            hbf, aggrbf, uwt[l], ub[l], tmp, Nn, 2 * Hh);
        ln_kernel<<<dim3((Nn + 7) / 8), b256, 0, stream>>>(
            tmp, ls[l], lb[l], h, hbf, Nn, 1);
    }

    // ---- pooling + classifier head ----
    pool_kernel<<<dim3(Gg), b256, 0, stream>>>(h, batch, gpool, Nn);
    fc_kernel<<<dim3((Gg * 256 + 255) / 256), b256, 0, stream>>>(
        gpool, cw1, cb1, g1, Gg, 512, 256, 1);
    fc_kernel<<<dim3((Gg * 128 + 255) / 256), b256, 0, stream>>>(
        g1, cw2, cb2, g2, Gg, 256, 128, 1);
    fc_kernel<<<dim3(1), b256, 0, stream>>>(
        g2, cw3, cb3, (float*)d_out, Gg, 128, 1, 0);
}
